// MultiHeadAttention_53944789238564
// MI455X (gfx1250) — matrix operand order repaired and checked by reading, not compiled, not run on hardware
//
#include <hip/hip_runtime.h>
#include <hip/hip_bf16.h>
#include <math.h>

// Problem constants (fixed by the reference).
constexpr int Bc  = 4;
constexpr int Sc  = 2048;
constexpr int Dc  = 1024;
constexpr int Hc  = 16;
constexpr int HDc = 64;

typedef __bf16 bf16_t;
typedef __attribute__((ext_vector_type(16))) __bf16 v16bf;
typedef __attribute__((ext_vector_type(8)))  __bf16 v8bf;
typedef __attribute__((ext_vector_type(8)))  float  v8f;

union V16 { v16bf v; v8bf h[2]; };

__device__ inline v8f wmma_bf16(v16bf a, v16bf b, v8f c) {
    return __builtin_amdgcn_wmma_f32_16x16x32_bf16(
        /*neg_a=*/false, a, /*neg_b=*/false, b,
        /*c_mod=*/(short)0, c, /*reuse_a=*/false, /*reuse_b=*/false);
}

// gfx1250 async DMA: 16B per lane, global -> LDS, tracked by ASYNCcnt.
// VDST is the wave-relative LDS byte address (low 32 bits of the flat
// pointer to __shared__), VADDR the 64-bit global address, no SADDR.
__device__ inline void async_copy_b128(const bf16_t* gptr, bf16_t* lptr) {
    unsigned lds_off = (unsigned)(uintptr_t)lptr;
    asm volatile("global_load_async_to_lds_b128 %0, %1, off"
                 :: "v"(lds_off), "v"(gptr)
                 : "memory");
}

__device__ inline float rmax16(float x) {
    #pragma unroll
    for (int off = 1; off < 16; off <<= 1) x = fmaxf(x, __shfl_xor(x, off, 32));
    return x;
}
__device__ inline float rsum16(float x) {
    #pragma unroll
    for (int off = 1; off < 16; off <<= 1) x += __shfl_xor(x, off, 32);
    return x;
}

// ---------------------------------------------------------------- cvt fp32->bf16
__global__ __launch_bounds__(256) void cvt_bf16_kernel(const float* __restrict__ in,
                                                       bf16_t* __restrict__ out, size_t n) {
    size_t i = ((size_t)blockIdx.x * 256 + threadIdx.x) * 4;
    if (i + 3 < n) {
        float4 f = *(const float4*)(in + i);
        out[i + 0] = (bf16_t)f.x;
        out[i + 1] = (bf16_t)f.y;
        out[i + 2] = (bf16_t)f.z;
        out[i + 3] = (bf16_t)f.w;
    }
}

// --------------------------------------------- transpose+cvt: W[K,N] -> Wt[N,K] bf16
__global__ __launch_bounds__(256) void transpose_cvt_kernel(const float* __restrict__ in,
                                                            bf16_t* __restrict__ out) {
    __shared__ float tile[16][17];
    int n0 = blockIdx.x * 16, k0 = blockIdx.y * 16;
    int tx = threadIdx.x % 16, ty = threadIdx.x / 16;
    tile[ty][tx] = in[(size_t)(k0 + ty) * Dc + n0 + tx];
    __syncthreads();
    out[(size_t)(n0 + ty) * Dc + k0 + tx] = (bf16_t)tile[tx][ty];
}

// ---------------------------------------------------------------- tiled WMMA GEMM
// C[M,N] = A[M,K] * Bt[N,K]^T + bias[N].
// Block tile 128(M) x 256(N); 8 waves as 2(M) x 4(N), each wave 64x64 =
// 4x4 WMMA accumulators (16 wmma per K=32 step).
// Staging is double-buffered via global_load_async_to_lds_b128: tile t+1
// streams into the other LDS buffer while tile t is consumed by WMMA.
#define LDT 40   // padded LDS stride (elems); 80B rows keep 16B alignment

__global__ __launch_bounds__(256) void gemm_bias_kernel(
    const bf16_t* __restrict__ A,   // [M, Kdim] row-major
    const bf16_t* __restrict__ Bt,  // [N, Kdim] (W transposed)
    const float*  __restrict__ bias,
    void* __restrict__ out,
    int M, int N, int Kdim, int mode)  // 0: bf16 [B,H,S,HD]; 1: bf16 [B,H,HD,S]; 2: fp32 [M,N]
{
    __shared__ bf16_t As[2][128 * LDT];   // 2 x 10 KB
    __shared__ bf16_t Bs[2][256 * LDT];   // 2 x 20 KB

    const int m0 = blockIdx.x * 128, n0 = blockIdx.y * 256;
    const int tid  = threadIdx.x;
    const int wave = tid >> 5, lane = tid & 31;
    const int wm = wave & 1, wn = wave >> 1;          // 2 waves in M, 4 in N
    const int lm = lane & 15, lhalf = lane >> 4;

    v8f acc[4][4] = {};

    // Stage one K=32 slab of A (128x32) and B (256x32) into LDS buffer `buf`.
    auto stage = [&](int buf, int k0) {
        #pragma unroll
        for (int it = 0; it < 2; ++it) {              // A: 512 16B chunks
            int c = tid + it * 256;
            int row = c >> 2, cc = c & 3;
            async_copy_b128(A + (size_t)(m0 + row) * Kdim + k0 + cc * 8,
                            &As[buf][row * LDT + cc * 8]);
        }
        #pragma unroll
        for (int it = 0; it < 4; ++it) {              // B: 1024 16B chunks
            int c = tid + it * 256;
            int row = c >> 2, cc = c & 3;
            async_copy_b128(Bt + (size_t)(n0 + row) * Kdim + k0 + cc * 8,
                            &Bs[buf][row * LDT + cc * 8]);
        }
    };

    stage(0, 0);
    const int nk = Kdim / 32;
    for (int t = 0; t < nk; ++t) {
        __builtin_amdgcn_s_wait_asynccnt(0);   // my DMA for tile t done
        __syncthreads();                       // everyone's DMA done / prev reads done
        if (t + 1 < nk) stage((t + 1) & 1, (t + 1) * 32);

        const bf16_t* Ab_ = As[t & 1];
        const bf16_t* Bb_ = Bs[t & 1];
        V16 afrag[4], bfrag[4];
        #pragma unroll
        for (int i = 0; i < 4; ++i) {
            const bf16_t* p = Ab_ + (wm * 64 + i * 16 + lm) * LDT;
            afrag[i].h[0] = *(const v8bf*)(p + lhalf * 8);
            afrag[i].h[1] = *(const v8bf*)(p + 16 + lhalf * 8);
        }
        #pragma unroll
        for (int j = 0; j < 4; ++j) {
            const bf16_t* p = Bb_ + (wn * 64 + j * 16 + lm) * LDT;
            bfrag[j].h[0] = *(const v8bf*)(p + lhalf * 8);
            bfrag[j].h[1] = *(const v8bf*)(p + 16 + lhalf * 8);
        }
        #pragma unroll
        for (int i = 0; i < 4; ++i)
            #pragma unroll
            for (int j = 0; j < 4; ++j)
                acc[i][j] = wmma_bf16(afrag[i].v, bfrag[j].v, acc[i][j]);
    }

    // Epilogue: C layout is (M = v + 8*lhalf, N = lm) per 16x16 tile.
    #pragma unroll
    for (int i = 0; i < 4; ++i) {
        #pragma unroll
        for (int j = 0; j < 4; ++j) {
            #pragma unroll
            for (int v = 0; v < 8; ++v) {
                int m = m0 + wm * 64 + i * 16 + v + 8 * lhalf;
                int n = n0 + wn * 64 + j * 16 + lm;
                float val = acc[i][j][v] + bias[n];
                if (mode == 2) {
                    ((float*)out)[(size_t)m * N + n] = val;
                } else {
                    int bI = m / Sc, sI = m % Sc, hI = n / HDc, hd = n % HDc;
                    size_t idx = (mode == 0)
                        ? (((size_t)(bI * Hc + hI) * Sc + sI) * HDc + hd)
                        : (((size_t)(bI * Hc + hI) * HDc + hd) * Sc + sI);
                    ((bf16_t*)out)[idx] = (bf16_t)val;
                }
            }
        }
    }
}

// ---------------------------------------------------------------- flash attention
// Q,K: [B,H,S,HD] bf16.  Vt: [B,H,HD,S] bf16.  Aout: [B*S, D] bf16 (cols = h*HD+hd).
// One wave owns 16 queries; 8 waves per block -> 128 queries per block.
__global__ __launch_bounds__(256) void flash_attn_kernel(
    const bf16_t* __restrict__ Q,
    const bf16_t* __restrict__ K,
    const bf16_t* __restrict__ Vt,
    bf16_t* __restrict__ Aout)
{
    const int h = blockIdx.y, b = blockIdx.z;
    const int wave = threadIdx.x >> 5, lane = threadIdx.x & 31;
    const int lm = lane & 15, lhalf = lane >> 4;
    const int qBase = blockIdx.x * 128 + wave * 16;

    const bf16_t* Qh = Q  + (size_t)(b * Hc + h) * Sc * HDc;
    const bf16_t* Kh = K  + (size_t)(b * Hc + h) * Sc * HDc;
    const bf16_t* Vh = Vt + (size_t)(b * Hc + h) * HDc * Sc;

    __shared__ bf16_t Psh[8 * 16 * 32];     // per-wave 16x32 P staging
    bf16_t* Pw = Psh + wave * (16 * 32);

    // Q fragments (HD=64 -> two K=32 steps), resident in registers.
    V16 qf[2];
    #pragma unroll
    for (int kk = 0; kk < 2; ++kk) {
        const bf16_t* qp = Qh + (size_t)(qBase + lm) * HDc + kk * 32;
        qf[kk].h[0] = *(const v8bf*)(qp + lhalf * 8);
        qf[kk].h[1] = *(const v8bf*)(qp + 16 + lhalf * 8);
    }

    v8f o[4] = {};
    float mrow[8], lrow[8];
    #pragma unroll
    for (int v = 0; v < 8; ++v) { mrow[v] = -INFINITY; lrow[v] = 0.0f; }

    const float scale = 0.125f;             // 1/sqrt(64)
    const int kEnd = qBase + 16;            // causal: keys < kEnd

    for (int k0 = 0; k0 < kEnd; k0 += 32) {
        // ---- scores: two 16-key subtiles, each 2 WMMA over HD
        v8f s[2] = {};
        #pragma unroll
        for (int sub = 0; sub < 2; ++sub) {
            int kb = k0 + sub * 16;
            int krow = kb + lm; if (krow > Sc - 1) krow = Sc - 1;
            #pragma unroll
            for (int kk = 0; kk < 2; ++kk) {
                V16 kf;
                const bf16_t* kp = Kh + (size_t)krow * HDc + kk * 32;
                kf.h[0] = *(const v8bf*)(kp + lhalf * 8);
                kf.h[1] = *(const v8bf*)(kp + 16 + lhalf * 8);
                s[sub] = wmma_bf16(qf[kk].v, kf.v, s[sub]);
            }
        }

        // ---- online softmax (row M = v + 8*lhalf lives across one 16-lane half)
        #pragma unroll
        for (int v = 0; v < 8; ++v) {
            int qI = qBase + v + 8 * lhalf;
            float s0 = s[0][v] * scale;
            float s1 = s[1][v] * scale;
            if (k0 + lm      > qI) s0 = -INFINITY;
            if (k0 + 16 + lm > qI) s1 = -INFINITY;
            float cm = rmax16(fmaxf(s0, s1));
            float mn = fmaxf(mrow[v], cm);
            float alpha = __expf(mrow[v] - mn);
            mrow[v] = mn;
            float p0 = __expf(s0 - mn);
            float p1 = __expf(s1 - mn);
            lrow[v] = lrow[v] * alpha + rsum16(p0 + p1);
            #pragma unroll
            for (int j = 0; j < 4; ++j) o[j][v] *= alpha;
            Pw[(v + 8 * lhalf) * 32 + lm]      = (bf16_t)p0;
            Pw[(v + 8 * lhalf) * 32 + 16 + lm] = (bf16_t)p1;
        }

        // ---- P (C-layout) -> A-layout fragment via per-wave LDS bounce
        V16 pf;
        pf.h[0] = *(const v8bf*)(Pw + lm * 32 + lhalf * 16);
        pf.h[1] = *(const v8bf*)(Pw + lm * 32 + lhalf * 16 + 8);

        // ---- O += P * V  (V^T rows give contiguous keys)
        int kb2 = k0 + lhalf * 16; if (kb2 > Sc - 16) kb2 = Sc - 16;
        #pragma unroll
        for (int j = 0; j < 4; ++j) {
            V16 vf;
            const bf16_t* vp = Vh + (size_t)(j * 16 + lm) * Sc + kb2;
            vf.h[0] = *(const v8bf*)(vp);
            vf.h[1] = *(const v8bf*)(vp + 8);
            o[j] = wmma_bf16(pf.v, vf.v, o[j]);
        }
    }

    // ---- normalize + store attn-out as [B*S, D] bf16
    #pragma unroll
    for (int v = 0; v < 8; ++v) {
        float inv = 1.0f / lrow[v];
        int row = qBase + v + 8 * lhalf;
        size_t base = ((size_t)b * Sc + row) * Dc + h * HDc;
        #pragma unroll
        for (int j = 0; j < 4; ++j)
            Aout[base + j * 16 + lm] = (bf16_t)(o[j][v] * inv);
    }
}

// ---------------------------------------------------------------- launcher
extern "C" void kernel_launch(void* const* d_in, const int* in_sizes, int n_in,
                              void* d_out, int out_size, void* d_ws, size_t ws_size,
                              hipStream_t stream) {
    const float* x  = (const float*)d_in[0];
    const float* wq = (const float*)d_in[1];
    const float* bq = (const float*)d_in[2];
    const float* wk = (const float*)d_in[3];
    const float* bk = (const float*)d_in[4];
    const float* wv = (const float*)d_in[5];
    const float* bv = (const float*)d_in[6];
    const float* wo = (const float*)d_in[7];
    const float* bo = (const float*)d_in[8];
    float* out = (float*)d_out;

    constexpr size_t NX = (size_t)Bc * Sc * Dc;   // 8388608
    constexpr size_t NW = (size_t)Dc * Dc;        // 1048576

    bf16_t* ws  = (bf16_t*)d_ws;
    bf16_t* xb  = ws;
    bf16_t* wqt = xb  + NX;
    bf16_t* wkt = wqt + NW;
    bf16_t* wvt = wkt + NW;
    bf16_t* wot = wvt + NW;
    bf16_t* Qb  = wot + NW;
    bf16_t* Kb  = Qb  + NX;
    bf16_t* Vtb = Kb  + NX;   // keeps clamped K-tile overreads inside d_ws
    bf16_t* Ab  = Vtb + NX;

    // 1) precision conversion
    cvt_bf16_kernel<<<dim3(NX / 1024), 256, 0, stream>>>(x, xb, NX);
    dim3 tg(Dc / 16, Dc / 16);
    transpose_cvt_kernel<<<tg, 256, 0, stream>>>(wq, wqt);
    transpose_cvt_kernel<<<tg, 256, 0, stream>>>(wk, wkt);
    transpose_cvt_kernel<<<tg, 256, 0, stream>>>(wv, wvt);
    transpose_cvt_kernel<<<tg, 256, 0, stream>>>(wo, wot);

    // 2) QKV projections (WMMA GEMMs, async double-buffered staging)
    dim3 gg(Bc * Sc / 128, Dc / 256);
    gemm_bias_kernel<<<gg, 256, 0, stream>>>(xb, wqt, bq, Qb,  Bc * Sc, Dc, Dc, 0);
    gemm_bias_kernel<<<gg, 256, 0, stream>>>(xb, wkt, bk, Kb,  Bc * Sc, Dc, Dc, 0);
    gemm_bias_kernel<<<gg, 256, 0, stream>>>(xb, wvt, bv, Vtb, Bc * Sc, Dc, Dc, 1);

    // 3) causal flash attention
    flash_attn_kernel<<<dim3(Sc / 128, Hc, Bc), 256, 0, stream>>>(Qb, Kb, Vtb, Ab);

    // 4) output projection -> fp32
    gemm_bias_kernel<<<gg, 256, 0, stream>>>(Ab, wot, bo, out, Bc * Sc, Dc, Dc, 2);
}
